// EfficientAttention_2422361555553
// MI455X (gfx1250) — matrix operand order repaired and checked by reading, not compiled, not run on hardware
//
#include <hip/hip_runtime.h>
#include <hip/hip_bf16.h>
#include <math.h>

// ---------------------------------------------------------------------------
// EfficientAttention for MI455X (gfx1250), wave32 + WMMA bf16.
// Shapes fixed by the reference: B=8, C=512, Cq=32, H=W=128, sh=sw=32, N=1024.
// All matmuls via v_wmma_f32_16x16x32_bf16 (f32 accum). Intermediates live in
// d_ws (~50 MiB, L2-resident on a 192 MB L2 part). The out-GEMM stages its
// shared B-operand tiles through LDS with GLOBAL_LOAD_ASYNC_TO_LDS_B128
// (double-buffered, ASYNCcnt-tracked).
// ---------------------------------------------------------------------------

#define BB   8
#define CC   512
#define CQ   32
#define HH   128
#define WW   128
#define SH   32
#define SW   32
#define NN   1024   // SH*SW tokens per batch

#if defined(__has_builtin)
#  if __has_builtin(__builtin_amdgcn_global_load_async_to_lds_b128) && \
      __has_builtin(__builtin_amdgcn_s_wait_asynccnt)
#    define USE_ASYNC_LDS 1
#  endif
#endif
#ifndef USE_ASYNC_LDS
#  define USE_ASYNC_LDS 0
#endif

typedef __attribute__((ext_vector_type(16))) __bf16 v16bf;
typedef __attribute__((ext_vector_type(8)))  float  v8f;
typedef int v4i __attribute__((ext_vector_type(4)));
typedef v4i __attribute__((address_space(1))) v4i_g;   // global v4i
typedef v4i __attribute__((address_space(3))) v4i_l;   // LDS v4i

struct U32x8 { unsigned int v[8]; };

__device__ __forceinline__ unsigned short f2bf(float f) {
  unsigned int u = __float_as_uint(f);
  u += 0x7fffu + ((u >> 16) & 1u);          // round-to-nearest-even
  return (unsigned short)(u >> 16);
}

// A-matrix 16x32 bf16 tile (M x K), src points at (row0, k0), ld in elements.
// ISA 7.12.2: lanes 0-15 hold M=lane, K={0..7,16..23}; lanes 16-31 K={8..15,24..31}.
__device__ __forceinline__ v16bf loadA(const unsigned short* __restrict__ src,
                                       int ld, int lane) {
  const int row = lane & 15;
  const int kb  = (lane >> 4) ? 8 : 0;
  U32x8 p;
#pragma unroll
  for (int j = 0; j < 8; ++j) {
    const int k = ((j < 4) ? 0 : 16) + kb + 2 * (j & 3);
    p.v[j] = *(const unsigned int*)(src + (size_t)row * ld + k);
  }
  return __builtin_bit_cast(v16bf, p);
}

// B-matrix 32x16 bf16 tile (K x N), src points at (k0, n0), ld in elements.
// Lane holds N=lane&15; lanes 0-15 K={0..7,16..23}, lanes 16-31 K={8..15,24..31}.
__device__ __forceinline__ v16bf loadB(const unsigned short* __restrict__ src,
                                       int ld, int lane) {
  const int n  = lane & 15;
  const int kb = (lane >> 4) ? 8 : 0;
  U32x8 p;
#pragma unroll
  for (int j = 0; j < 8; ++j) {
    const int k = ((j < 4) ? 0 : 16) + kb + 2 * (j & 3);
    p.v[j] = (unsigned int)src[(size_t)k * ld + n] |
             ((unsigned int)src[(size_t)(k + 1) * ld + n] << 16);
  }
  return __builtin_bit_cast(v16bf, p);
}

// B tile read out of an LDS staging buffer holding a packed 32x16 bf16 tile
// (row-major, 8 dwords per K-row).
__device__ __forceinline__ v16bf loadB_lds(const unsigned int* s, int lane) {
  const unsigned short* h = (const unsigned short*)s;
  const int n  = lane & 15;
  const int kb = (lane >> 4) ? 8 : 0;
  U32x8 p;
#pragma unroll
  for (int j = 0; j < 8; ++j) {
    const int k = ((j < 4) ? 0 : 16) + kb + 2 * (j & 3);
    p.v[j] = (unsigned int)h[k * 16 + n] | ((unsigned int)h[(k + 1) * 16 + n] << 16);
  }
  return __builtin_bit_cast(v16bf, p);
}

__device__ __forceinline__ v8f wmma_bf16(v16bf a, v16bf b, v8f c) {
  // (neg_a, A, neg_b, B, c_mod, C, reuse_a, reuse_b)
  return __builtin_amdgcn_wmma_f32_16x16x32_bf16(false, a, false, b,
                                                 (short)0, c, false, false);
}

// --------------------------- elementwise kernels ---------------------------

__global__ __launch_bounds__(256) void cvt_bf16_kernel(
    const float* __restrict__ src, unsigned short* __restrict__ dst, int n) {
  int i = blockIdx.x * 256 + threadIdx.x;
  if (i < n) dst[i] = f2bf(src[i]);
}

// Bilinear downsample 128x128 -> 32x32 (+ fp32 -> bf16). xs layout [b][c][n].
__global__ __launch_bounds__(256) void downsample_kernel(
    const float* __restrict__ x, unsigned short* __restrict__ xs) {
  int e = blockIdx.x * 256 + threadIdx.x;       // BB*CC*NN = 4,194,304
  if (e >= BB * CC * NN) return;
  int n  = e & (NN - 1);
  int c  = (e >> 10) & (CC - 1);
  int b  = e >> 19;
  int oy = n >> 5, ox = n & 31;

  float fy = (oy + 0.5f) * ((float)HH / SH) - 0.5f;
  fy = fminf(fmaxf(fy, 0.0f), (float)(HH - 1));
  int y0 = (int)floorf(fy); int y1 = min(y0 + 1, HH - 1); float wy = fy - y0;

  float fx = (ox + 0.5f) * ((float)WW / SW) - 0.5f;
  fx = fminf(fmaxf(fx, 0.0f), (float)(WW - 1));
  int x0 = (int)floorf(fx); int x1 = min(x0 + 1, WW - 1); float wx = fx - x0;

  const float* p = x + ((size_t)b * CC + c) * (HH * WW);
  float v00 = p[y0 * WW + x0], v01 = p[y0 * WW + x1];
  float v10 = p[y1 * WW + x0], v11 = p[y1 * WW + x1];
  float top = v00 + (v01 - v00) * wx;
  float bot = v10 + (v11 - v10) * wx;
  xs[e] = f2bf(top + (bot - top) * wy);
}

// Bilinear upsample 32x32 -> 128x128, gamma*up + x residual (fp32 out).
__global__ __launch_bounds__(256) void upsample_residual_kernel(
    const float* __restrict__ outs,   // [b][c][NN] fp32
    const float* __restrict__ x, const float* __restrict__ gamma,
    float* __restrict__ out) {
  size_t e = (size_t)blockIdx.x * 256 + threadIdx.x;  // 67,108,864
  if (e >= (size_t)BB * CC * HH * WW) return;
  int px = (int)(e & (WW - 1));
  int py = (int)((e >> 7) & (HH - 1));
  int c  = (int)((e >> 14) & (CC - 1));
  int b  = (int)(e >> 23);

  float fy = (py + 0.5f) * ((float)SH / HH) - 0.5f;
  fy = fminf(fmaxf(fy, 0.0f), (float)(SH - 1));
  int y0 = (int)floorf(fy); int y1 = min(y0 + 1, SH - 1); float wy = fy - y0;

  float fx = (px + 0.5f) * ((float)SW / WW) - 0.5f;
  fx = fminf(fmaxf(fx, 0.0f), (float)(SW - 1));
  int x0 = (int)floorf(fx); int x1 = min(x0 + 1, SW - 1); float wx = fx - x0;

  const float* p = outs + ((size_t)b * CC + c) * NN;
  float v00 = p[y0 * SW + x0], v01 = p[y0 * SW + x1];
  float v10 = p[y1 * SW + x0], v11 = p[y1 * SW + x1];
  float top = v00 + (v01 - v00) * wx;
  float bot = v10 + (v11 - v10) * wx;
  out[e] = gamma[0] * (top + (bot - top) * wy) + x[e];
}

// ------------------------------ WMMA kernels -------------------------------

// D[m,n] = sum_k Wbf[m,k] * X[b][k,n] + bias[m]; bf16 out.
// transposeOut=1 stores D transposed ([b][n][M]) for use as a later A operand.
// grid = (NN/128, M/16, BB), block = 256 (8 waves, one 16x16 tile per wave).
__global__ __launch_bounds__(256) void proj_gemm_kernel(
    const unsigned short* __restrict__ Wbf, const float* __restrict__ bias,
    const unsigned short* __restrict__ Xall, unsigned short* __restrict__ Out,
    int M, int Kdim, int transposeOut) {
  const int lane = threadIdx.x & 31;
  const int wid  = threadIdx.x >> 5;
  const int n0   = (blockIdx.x * 8 + wid) * 16;
  const int m0   = blockIdx.y * 16;
  const int b    = blockIdx.z;
  const unsigned short* X = Xall + (size_t)b * Kdim * NN;

  v8f acc = {0.f, 0.f, 0.f, 0.f, 0.f, 0.f, 0.f, 0.f};
  for (int k0 = 0; k0 < Kdim; k0 += 32) {
    v16bf a  = loadA(Wbf + (size_t)m0 * Kdim + k0, Kdim, lane);
    v16bf bt = loadB(X + (size_t)k0 * NN + n0, NN, lane);
    acc = wmma_bf16(a, bt, acc);
  }
  const int nn  = n0 + (lane & 15);
  const int mhi = (lane >> 4) * 8;
#pragma unroll
  for (int i = 0; i < 8; ++i) {
    int mm = m0 + i + mhi;
    unsigned short v = f2bf(acc[i] + bias[mm]);
    if (transposeOut) Out[(size_t)b * NN * M + (size_t)nn * M + mm] = v;
    else              Out[(size_t)b * M * NN + (size_t)mm * NN + nn] = v;
  }
}

// Fused scores + softmax. One block = 16 query rows x all 1024 keys.
// scores[m,n] = (qT[m,:] . k[:,n]) / sqrt(32)  -> one WMMA per tile (K=32).
// attn is written TRANSPOSED ([b][n][m]) so the out-GEMM B operand is a
// contiguous row-major K x N tile.
// grid = (NN/16, BB), block = 256 (8 waves x 8 column-tiles each).
__global__ __launch_bounds__(256) void scores_softmax_kernel(
    const unsigned short* __restrict__ qT,   // [b][NN][CQ]
    const unsigned short* __restrict__ kmat, // [b][CQ][NN]
    unsigned short* __restrict__ attnT) {    // [b][NN(n)][NN(m)] bf16
  __shared__ float sc[16 * NN];              // 64 KB
  const int tid  = threadIdx.x;
  const int lane = tid & 31;
  const int wid  = tid >> 5;
  const int m0   = blockIdx.x * 16;
  const int b    = blockIdx.y;
  const unsigned short* qTb = qT   + (size_t)b * NN * CQ;
  const unsigned short* kb  = kmat + (size_t)b * CQ * NN;

  v16bf a = loadA(qTb + (size_t)m0 * CQ, CQ, lane);   // shared across tiles
  const float scale = 0.17677669529663687f;           // 1/sqrt(32)
#pragma unroll
  for (int t = 0; t < 8; ++t) {
    const int n0 = (wid * 8 + t) * 16;
    v16bf bt = loadB(kb + n0, NN, lane);
    v8f acc = {0.f, 0.f, 0.f, 0.f, 0.f, 0.f, 0.f, 0.f};
    acc = wmma_bf16(a, bt, acc);
    const int col = n0 + (lane & 15);
    const int rhi = (lane >> 4) * 8;
#pragma unroll
    for (int i = 0; i < 8; ++i) sc[(i + rhi) * NN + col] = acc[i] * scale;
  }
  __syncthreads();

  // Row softmax: 16 threads per row; reductions via 16-lane shuffles.
  const int r  = tid >> 4;    // 0..15 (two rows per wave: lanes 0-15 / 16-31)
  const int cl = tid & 15;
  float mx = -3.402823466e38f;
  for (int c = cl; c < NN; c += 16) mx = fmaxf(mx, sc[r * NN + c]);
#pragma unroll
  for (int off = 8; off >= 1; off >>= 1) mx = fmaxf(mx, __shfl_xor(mx, off, 16));
  float s = 0.f;
  for (int c = cl; c < NN; c += 16) {
    float e = __expf(sc[r * NN + c] - mx);
    sc[r * NN + c] = e;
    s += e;
  }
#pragma unroll
  for (int off = 8; off >= 1; off >>= 1) s += __shfl_xor(s, off, 16);
  const float inv = 1.0f / s;
  unsigned short* aT = attnT + (size_t)b * NN * NN + (m0 + r);  // column m
  for (int c = cl; c < NN; c += 16)
    aT[(size_t)c * NN] = f2bf(sc[r * NN + c] * inv);
}

// out_small[b][cout][m] = sum_n v[b][cout][n] * attnT[b][n][m]   (fp32 out)
// All 8 waves in a block share the same m-tile, so the B operand tile
// (attnT[k0..k0+31][m0..m0+15]) is identical across waves: each wave
// double-buffers it through its own LDS slot with async-to-LDS loads.
// grid = (NN/16, CC/128, BB), block = 256; K = NN -> 32 WMMAs per wave.
__global__ __launch_bounds__(256) void out_gemm_kernel(
    const unsigned short* __restrict__ vmat,   // [b][CC][NN]
    const unsigned short* __restrict__ attnT,  // [b][NN(n)][NN(m)]
    float* __restrict__ outs) {                // [b][CC][NN]
  const int lane = threadIdx.x & 31;
  const int wid  = threadIdx.x >> 5;
  const int m0   = blockIdx.x * 16;               // token tile (shared)
  const int c0   = blockIdx.y * 128 + wid * 16;   // channel tile (per wave)
  const int b    = blockIdx.z;
  const unsigned short* vb = vmat  + (size_t)b * CC * NN;
  const unsigned short* aT = attnT + (size_t)b * NN * NN;

  v8f acc = {0.f, 0.f, 0.f, 0.f, 0.f, 0.f, 0.f, 0.f};

#if USE_ASYNC_LDS
  __shared__ unsigned int stage[2][8][32 * 8];    // 2 bufs x 8 waves x 1 KB
  // Each lane DMAs one 32-byte K-row of the tile into LDS (2x b128).
  auto issue = [&](int buf, int k0) {
    unsigned short* g =
        const_cast<unsigned short*>(aT + (size_t)(k0 + lane) * NN + m0);
    unsigned int* l = &stage[buf][wid][lane * 8];
    __builtin_amdgcn_global_load_async_to_lds_b128(
        (v4i_g*)g, (v4i_l*)l, 0, 0);
    __builtin_amdgcn_global_load_async_to_lds_b128(
        (v4i_g*)(g + 8), (v4i_l*)(l + 4), 0, 0);
  };
  auto compute = [&](int buf, int k0) {
    v16bf bt = loadB_lds(&stage[buf][wid][0], lane);
    v16bf a  = loadA(vb + (size_t)c0 * NN + k0, NN, lane);
    acc = wmma_bf16(a, bt, acc);
  };
  issue(0, 0);
  issue(1, 32);
  int k0 = 0;
  for (; k0 < NN - 64; k0 += 32) {
    const int buf = (k0 >> 5) & 1;
    __builtin_amdgcn_s_wait_asynccnt(2);   // current tile landed in LDS
    compute(buf, k0);
    issue(buf, k0 + 64);                   // refill (ds reads already consumed
  }                                        //  by the WMMA's dscnt wait)
  __builtin_amdgcn_s_wait_asynccnt(2);
  compute((k0 >> 5) & 1, k0);
  k0 += 32;
  __builtin_amdgcn_s_wait_asynccnt(0);
  compute((k0 >> 5) & 1, k0);
#else
  for (int k0 = 0; k0 < NN; k0 += 32) {
    v16bf a  = loadA(vb + (size_t)c0 * NN + k0, NN, lane);
    v16bf bt = loadB(aT + (size_t)k0 * NN + m0, NN, lane);
    acc = wmma_bf16(a, bt, acc);
  }
#endif

  const int mm  = m0 + (lane & 15);
  const int chi = (lane >> 4) * 8;
#pragma unroll
  for (int i = 0; i < 8; ++i)
    outs[((size_t)b * CC + c0 + i + chi) * NN + mm] = acc[i];
}

// ------------------------------- launcher ----------------------------------

extern "C" void kernel_launch(void* const* d_in, const int* in_sizes, int n_in,
                              void* d_out, int out_size, void* d_ws, size_t ws_size,
                              hipStream_t stream) {
  const float* x     = (const float*)d_in[0];
  const float* wq    = (const float*)d_in[1];
  const float* bq    = (const float*)d_in[2];
  const float* wk    = (const float*)d_in[3];
  const float* bk    = (const float*)d_in[4];
  const float* wv    = (const float*)d_in[5];
  const float* bv    = (const float*)d_in[6];
  const float* gamma = (const float*)d_in[7];
  float* out = (float*)d_out;

  unsigned char* ws = (unsigned char*)d_ws;
  size_t off = 0;
  auto alloc = [&](size_t bytes) {
    size_t o = off;
    off = (off + bytes + 255) & ~(size_t)255;
    return o;
  };
  unsigned short* xs_bf = (unsigned short*)(ws + alloc((size_t)BB * CC * NN * 2)); // 8 MiB
  unsigned short* wq_bf = (unsigned short*)(ws + alloc((size_t)CQ * CC * 2));
  unsigned short* wk_bf = (unsigned short*)(ws + alloc((size_t)CQ * CC * 2));
  unsigned short* wv_bf = (unsigned short*)(ws + alloc((size_t)CC * CC * 2));
  unsigned short* qT_bf = (unsigned short*)(ws + alloc((size_t)BB * NN * CQ * 2));
  unsigned short* k_bf  = (unsigned short*)(ws + alloc((size_t)BB * CQ * NN * 2));
  unsigned short* v_bf  = (unsigned short*)(ws + alloc((size_t)BB * CC * NN * 2)); // 8 MiB
  unsigned short* attnT = (unsigned short*)(ws + alloc((size_t)BB * NN * NN * 2)); // 16 MiB
  float*          outs  = (float*)        (ws + alloc((size_t)BB * CC * NN * 4)); // 16 MiB
  (void)ws_size; (void)in_sizes; (void)n_in; (void)out_size;

  // Weights -> bf16
  cvt_bf16_kernel<<<(CQ * CC + 255) / 256, 256, 0, stream>>>(wq, wq_bf, CQ * CC);
  cvt_bf16_kernel<<<(CQ * CC + 255) / 256, 256, 0, stream>>>(wk, wk_bf, CQ * CC);
  cvt_bf16_kernel<<<(CC * CC + 255) / 256, 256, 0, stream>>>(wv, wv_bf, CC * CC);

  // Downsample + bf16
  downsample_kernel<<<(BB * CC * NN) / 256, 256, 0, stream>>>(x, xs_bf);

  // Projections (q stored transposed for the scores A-operand)
  proj_gemm_kernel<<<dim3(NN / 128, CQ / 16, BB), 256, 0, stream>>>(
      wq_bf, bq, xs_bf, qT_bf, CQ, CC, 1);
  proj_gemm_kernel<<<dim3(NN / 128, CQ / 16, BB), 256, 0, stream>>>(
      wk_bf, bk, xs_bf, k_bf, CQ, CC, 0);
  proj_gemm_kernel<<<dim3(NN / 128, CC / 16, BB), 256, 0, stream>>>(
      wv_bf, bv, xs_bf, v_bf, CC, CC, 0);

  // Fused scores + softmax (writes attn transposed)
  scores_softmax_kernel<<<dim3(NN / 16, BB), 256, 0, stream>>>(qT_bf, k_bf, attnT);

  // out_small = V x attn^T (async-LDS double-buffered B operand)
  out_gemm_kernel<<<dim3(NN / 16, CC / 128, BB), 256, 0, stream>>>(v_bf, attnT, outs);

  // Upsample + residual
  upsample_residual_kernel<<<(unsigned)(((size_t)BB * CC * HH * WW) / 256), 256, 0,
                             stream>>>(outs, x, gamma, out);
}
